// MultiHeadSelfAttention_88691074662954
// MI455X (gfx1250) — compile-verified, matrix operand order checked by reading, not run on hardware
//
#include <hip/hip_runtime.h>
#include <hip/hip_bf16.h>

// MI455X (gfx1250) multi-head causal self-attention, flash-attention style.
// All matmuls go through v_wmma_f32_16x16x32_f16 (f16 operands, f32 accum);
// softmax statistics stay in f32. Attention waves own 32 query rows (two
// 16x16 C tiles) so each K/V fragment load feeds 2x the WMMAs.

typedef __attribute__((ext_vector_type(16))) _Float16 v16h;
typedef __attribute__((ext_vector_type(8)))  float    v8f;

#define DEV __device__ __forceinline__

// ---- problem constants -----------------------------------------------------
#define BATCH 2
#define SEQ   2048
#define HEADS 12
#define DHEAD 64
#define ROWS  (BATCH * SEQ)        // 4096
#define QKVC  (3 * HEADS * DHEAD)  // 2304
#define HD    (HEADS * DHEAD)      // 768
#define WPB   4                    // waves per block (128 threads)

// ---- WMMA fragment index math (ISA 7.12.2, wave32) -------------------------
// A fragment (16xK, f16): lane holds row (lane&15); element i maps to
//   K = (i&7) + 16*(i>>3) + 8*(lane>>4)
// Runs of 8 consecutive i are contiguous in K, so row-major gathers lower to
// 2x global_load_b128 per lane (verified in round-1 asm).
DEV int frag_k(int i, int lane) {
  return (i & 7) + ((i >> 3) << 4) + ((lane >> 4) << 3);
}

// A-fragment from row-major f16 matrix (also serves as B-fragment of its
// transpose, i.e. for S = Q*K^T we load K row-major with this helper).
DEV v16h load_a_f16(const _Float16* __restrict__ p, int ld, int row0, int col0, int lane) {
  v16h a;
  const int r = row0 + (lane & 15);
#pragma unroll
  for (int i = 0; i < 16; ++i) a[i] = p[(size_t)r * ld + col0 + frag_k(i, lane)];
  return a;
}

DEV v16h load_a_f32(const float* __restrict__ p, int ld, int row0, int col0, int lane) {
  v16h a;
  const int r = row0 + (lane & 15);
#pragma unroll
  for (int i = 0; i < 16; ++i) a[i] = (_Float16)p[(size_t)r * ld + col0 + frag_k(i, lane)];
  return a;
}

// Dense B-fragment (KxN row-major): lane holds column (lane&15), K via frag_k.
DEV v16h load_b_f16(const _Float16* __restrict__ p, int ld, int k0, int col0, int lane) {
  v16h b;
  const int c = col0 + (lane & 15);
#pragma unroll
  for (int i = 0; i < 16; ++i) b[i] = p[(size_t)(k0 + frag_k(i, lane)) * ld + c];
  return b;
}

DEV v16h load_b_f32(const float* __restrict__ p, int ld, int k0, int col0, int lane) {
  v16h b;
  const int c = col0 + (lane & 15);
#pragma unroll
  for (int i = 0; i < 16; ++i) b[i] = (_Float16)p[(size_t)(k0 + frag_k(i, lane)) * ld + c];
  return b;
}

DEV v8f wmma16(v16h a, v16h b, v8f c) {
  // (neg_a, A, neg_b, B, c_mod, C, reuse_a, reuse_b)
  return __builtin_amdgcn_wmma_f32_16x16x32_f16(false, a, false, b, (short)0, c,
                                                false, false);
}

// ============================================================================
// Kernel 1: qkv = x @ Wqkv, scattered into f16 Q/K/V[h][b][s][d]
// ============================================================================
__global__ __launch_bounds__(WPB * 32)
void qkv_proj_kernel(const float* __restrict__ x, const float* __restrict__ W,
                     _Float16* __restrict__ Qw, _Float16* __restrict__ Kw,
                     _Float16* __restrict__ Vw) {
  const int lane = threadIdx.x & 31;
  const int wid  = blockIdx.x * WPB + (threadIdx.x >> 5);
  const int colTiles = QKVC / 16;  // 144
  const int rowTiles = ROWS / 16;  // 256
  if (wid >= rowTiles * colTiles) return;
  const int r0 = (wid / colTiles) * 16;
  const int c0 = (wid % colTiles) * 16;

  v8f acc = {};
#pragma unroll
  for (int k0 = 0; k0 < DHEAD; k0 += 32) {
    v16h a = load_a_f32(x, DHEAD, r0, k0, lane);
    v16h b = load_b_f32(W, QKVC, k0, c0, lane);
    acc = wmma16(a, b, acc);
  }

  // column block -> (head h, tensor t in {q,k,v}, d0); 16 | 64 so uniform per tile
  const int h  = c0 / (3 * DHEAD);
  const int t  = (c0 % (3 * DHEAD)) / DHEAD;
  const int d0 = c0 % DHEAD;
  const int bb = r0 / SEQ;
  const int s0 = r0 % SEQ;
  _Float16* dst = (t == 0) ? Qw : (t == 1) ? Kw : Vw;
  dst += (size_t)(h * BATCH + bb) * SEQ * DHEAD;
  const int col   = d0 + (lane & 15);
  const int rbase = (lane >> 4) << 3;
#pragma unroll
  for (int v = 0; v < 8; ++v)
    dst[(size_t)(s0 + v + rbase) * DHEAD + col] = (_Float16)acc[v];
}

// ============================================================================
// Kernel 2: causal flash attention. One wave = 32 query rows (two 16-row
// C tiles) of one (h,b); K/V fragments are loaded once per 32-key step and
// reused by both query sub-tiles (16 WMMAs per 8 fragment loads).
// ============================================================================
__global__ __launch_bounds__(WPB * 32)
void attn_kernel(const _Float16* __restrict__ Qw, const _Float16* __restrict__ Kw,
                 const _Float16* __restrict__ Vw, _Float16* __restrict__ Ow) {
  __shared__ _Float16 ldsP[WPB][2][16 * 32];  // per-wave P staging tiles

  const int lane  = threadIdx.x & 31;
  const int wslot = threadIdx.x >> 5;
  const int wid   = blockIdx.x * WPB + wslot;
  const int qTiles = SEQ / 32;  // 64
  if (wid >= HEADS * BATCH * qTiles) return;
  const int h  = wid / (BATCH * qTiles);
  const int bb = (wid % (BATCH * qTiles)) / qTiles;
  const int q0 = (wid % qTiles) * 32;

  const size_t hbo = (size_t)(h * BATCH + bb) * SEQ * DHEAD;
  const _Float16* Q = Qw + hbo;
  const _Float16* K = Kw + hbo;
  const _Float16* V = Vw + hbo;

  // Q fragments (2 query sub-tiles x 2 D k-steps), reused across the key loop
  v16h aQ[2][2];
#pragma unroll
  for (int sq = 0; sq < 2; ++sq) {
    aQ[sq][0] = load_a_f16(Q, DHEAD, q0 + 16 * sq, 0, lane);
    aQ[sq][1] = load_a_f16(Q, DHEAD, q0 + 16 * sq, 32, lane);
  }

  v8f o[2][4];
  float mrow[2][8], lrow[2][8];
#pragma unroll
  for (int sq = 0; sq < 2; ++sq)
#pragma unroll
    for (int v = 0; v < 8; ++v) {
      mrow[sq][v] = -3.0e38f;
      lrow[sq][v] = 0.0f;
      if (v < 4) o[sq][v] = (v8f){};
    }
#pragma unroll
  for (int sq = 0; sq < 2; ++sq)
#pragma unroll
    for (int n = 0; n < 4; ++n) o[sq][n] = (v8f){};

  const int cn    = lane & 15;
  const int rbase = (lane >> 4) << 3;
  const int nkt   = q0 / 32 + 1;  // causal: keys [0, q0+32)

  for (int kt = 0; kt < nkt; ++kt) {
    const int k0 = kt * 32;
    if (kt + 1 < nkt) {  // stream next K/V tiles toward L2/L0
      __builtin_prefetch(&K[(size_t)(k0 + 32) * DHEAD], 0, 1);
      __builtin_prefetch(&V[(size_t)(k0 + 32) * DHEAD], 0, 1);
    }

    // ---- shared K^T and V fragments for this 32-key step ----
    const v16h bk00 = load_a_f16(K, DHEAD, k0,      0,  lane);  // B = K^T
    const v16h bk01 = load_a_f16(K, DHEAD, k0,      32, lane);
    const v16h bk10 = load_a_f16(K, DHEAD, k0 + 16, 0,  lane);
    const v16h bk11 = load_a_f16(K, DHEAD, k0 + 16, 32, lane);
    const v16h bv0  = load_b_f16(V, DHEAD, k0, 0,  lane);
    const v16h bv1  = load_b_f16(V, DHEAD, k0, 16, lane);
    const v16h bv2  = load_b_f16(V, DHEAD, k0, 32, lane);
    const v16h bv3  = load_b_f16(V, DHEAD, k0, 48, lane);

#pragma unroll
    for (int sq = 0; sq < 2; ++sq) {
      // ---- S = (Q K^T) / sqrt(D), two 16-key subtiles ----
      v8f s0 = {}, s1 = {};
      s0 = wmma16(aQ[sq][0], bk00, s0);
      s0 = wmma16(aQ[sq][1], bk01, s0);
      s1 = wmma16(aQ[sq][0], bk10, s1);
      s1 = wmma16(aQ[sq][1], bk11, s1);

      // ---- scale + causal mask + online softmax ----
      float alpha[8];
#pragma unroll
      for (int v = 0; v < 8; ++v) {
        const int qrow = q0 + 16 * sq + v + rbase;
        float x0 = s0[v] * 0.125f;
        float x1 = s1[v] * 0.125f;
        if (k0 + cn > qrow)      x0 = -1.0e30f;
        if (k0 + 16 + cn > qrow) x1 = -1.0e30f;

        float mx = fmaxf(x0, x1);
#pragma unroll
        for (int off = 1; off < 16; off <<= 1)
          mx = fmaxf(mx, __shfl_xor(mx, off, 32));  // row lives in a 16-lane half
        const float mnew = fmaxf(mrow[sq][v], mx);
        const float a    = __expf(mrow[sq][v] - mnew);
        const float p0   = __expf(x0 - mnew);
        const float p1   = __expf(x1 - mnew);
        float rs = p0 + p1;
#pragma unroll
        for (int off = 1; off < 16; off <<= 1)
          rs += __shfl_xor(rs, off, 32);
        lrow[sq][v]  = lrow[sq][v] * a + rs;
        mrow[sq][v]  = mnew;
        alpha[v]     = a;
        s0[v] = p0;
        s1[v] = p1;
      }
#pragma unroll
      for (int v = 0; v < 8; ++v) {
        o[sq][0][v] *= alpha[v];
        o[sq][1][v] *= alpha[v];
        o[sq][2][v] *= alpha[v];
        o[sq][3][v] *= alpha[v];
      }

      // ---- restripe P: C-layout -> LDS (16x32 f16) ----
      _Float16* lp = ldsP[wslot][sq];
#pragma unroll
      for (int v = 0; v < 8; ++v) {
        const int r = v + rbase;
        lp[r * 32 + cn]      = (_Float16)s0[v];
        lp[r * 32 + 16 + cn] = (_Float16)s1[v];
      }
    }

    asm volatile("s_wait_dscnt 0" ::: "memory");  // same-wave DS RAW fence

    // ---- O += P @ V for both query sub-tiles (8 WMMAs on shared bv*) ----
#pragma unroll
    for (int sq = 0; sq < 2; ++sq) {
      const v16h aP = load_a_f16(ldsP[wslot][sq], 32, 0, 0, lane);
      o[sq][0] = wmma16(aP, bv0, o[sq][0]);
      o[sq][1] = wmma16(aP, bv1, o[sq][1]);
      o[sq][2] = wmma16(aP, bv2, o[sq][2]);
      o[sq][3] = wmma16(aP, bv3, o[sq][3]);
    }
  }

  // ---- normalize, store O into (B*S, H*D) f16 for the output projection ----
#pragma unroll
  for (int sq = 0; sq < 2; ++sq)
#pragma unroll
    for (int v = 0; v < 8; ++v) {
      const float inv = 1.0f / lrow[sq][v];
      const size_t row = (size_t)bb * SEQ + q0 + 16 * sq + v + rbase;
      _Float16* dst = Ow + row * HD + h * DHEAD;
      dst[cn]      = (_Float16)(o[sq][0][v] * inv);
      dst[16 + cn] = (_Float16)(o[sq][1][v] * inv);
      dst[32 + cn] = (_Float16)(o[sq][2][v] * inv);
      dst[48 + cn] = (_Float16)(o[sq][3][v] * inv);
    }
}

// ============================================================================
// Kernel 3: out = O @ Wout + bout  (4096x768 @ 768x64, fp32 result)
// ============================================================================
__global__ __launch_bounds__(WPB * 32)
void out_proj_kernel(const _Float16* __restrict__ Ow, const float* __restrict__ Wout,
                     const float* __restrict__ bout, float* __restrict__ out) {
  const int lane = threadIdx.x & 31;
  const int wid  = blockIdx.x * WPB + (threadIdx.x >> 5);
  const int rowTiles = ROWS / 16;   // 256
  const int colTiles = DHEAD / 16;  // 4
  if (wid >= rowTiles * colTiles) return;
  const int r0 = (wid / colTiles) * 16;
  const int c0 = (wid % colTiles) * 16;

  v8f acc = {};
#pragma unroll 4
  for (int k0 = 0; k0 < HD; k0 += 32) {
    v16h a = load_a_f16(Ow, HD, r0, k0, lane);
    v16h b = load_b_f32(Wout, DHEAD, k0, c0, lane);
    acc = wmma16(a, b, acc);
  }

  const int cn    = lane & 15;
  const int rbase = (lane >> 4) << 3;
  const float bias = bout[c0 + cn];
#pragma unroll
  for (int v = 0; v < 8; ++v)
    out[(size_t)(r0 + v + rbase) * DHEAD + c0 + cn] = acc[v] + bias;
}

// ============================================================================
// Host-side launcher
// ============================================================================
extern "C" void kernel_launch(void* const* d_in, const int* in_sizes, int n_in,
                              void* d_out, int out_size, void* d_ws, size_t ws_size,
                              hipStream_t stream) {
  const float* x    = (const float*)d_in[0];
  // d_in[1] = bool causal mask (tril) — causality computed analytically.
  const float* Wqkv = (const float*)d_in[2];
  const float* Wout = (const float*)d_in[3];
  const float* bout = (const float*)d_in[4];
  float* out = (float*)d_out;

  const size_t perTensor = (size_t)HEADS * BATCH * SEQ * DHEAD;  // 3,145,728 halves
  _Float16* Qw = (_Float16*)d_ws;
  _Float16* Kw = Qw + perTensor;
  _Float16* Vw = Kw + perTensor;
  _Float16* Ow = Vw + perTensor;  // 4096*768 halves

  // 1) QKV projection: 256 * 144 wave-tiles
  {
    const int tiles = (ROWS / 16) * (QKVC / 16);
    const int blocks = (tiles + WPB - 1) / WPB;
    qkv_proj_kernel<<<blocks, WPB * 32, 0, stream>>>(x, Wqkv, Qw, Kw, Vw);
  }
  // 2) Flash attention: 12 * 2 * 64 wave-tiles (32 query rows each)
  {
    const int tiles = HEADS * BATCH * (SEQ / 32);
    const int blocks = (tiles + WPB - 1) / WPB;
    attn_kernel<<<blocks, WPB * 32, 0, stream>>>(Qw, Kw, Vw, Ow);
  }
  // 3) Output projection: 256 * 4 wave-tiles
  {
    const int tiles = (ROWS / 16) * (DHEAD / 16);
    const int blocks = (tiles + WPB - 1) / WPB;
    out_proj_kernel<<<blocks, WPB * 32, 0, stream>>>(Ow, Wout, bout, out);
  }
}
